// GraphAttention_75574244540920
// MI455X (gfx1250) — matrix operand order repaired and checked by reading, not compiled, not run on hardware
//
#include <hip/hip_runtime.h>
#include <stdint.h>

// ---------------------------------------------------------------------------
// Problem constants
// ---------------------------------------------------------------------------
#define B_  8
#define N_  1024
#define C_  64
#define H_  8
#define ROWS_ (B_ * N_)             // 8192
#define SCALE_ (0.125f)             // C^-0.5

typedef __attribute__((ext_vector_type(16))) __bf16 v16bf;
typedef __attribute__((ext_vector_type(8)))  float  v8f;
typedef unsigned int u32x4 __attribute__((ext_vector_type(4)));
typedef int          i32x4 __attribute__((ext_vector_type(4)));
typedef int          i32x8 __attribute__((ext_vector_type(8)));
typedef int          v4i_vs __attribute__((vector_size(16)));   // matches builtin param

// ---------------------------------------------------------------------------
// gfx1250 data-movement feature detection (compile-safe layering)
// ---------------------------------------------------------------------------
#if __has_builtin(__builtin_amdgcn_tensor_load_to_lds)
#define HAVE_TDM 1
#else
#define HAVE_TDM 0
#endif

#if __has_builtin(__builtin_amdgcn_global_load_async_to_lds_b128)
#define HAVE_ASYNC 1
#else
#define HAVE_ASYNC 0
#endif

__device__ __forceinline__ uint32_t lds_off_of(const void* p) {
  return (uint32_t)(uintptr_t)(__attribute__((address_space(3))) const void*)p;
}

#if HAVE_TDM
// Tensor DMA descriptor per CDNA5 ISA §8.3/8.4.
// data_size = 8 bytes (code 3); dims/strides in 8-byte units.
// tile_d1 == 0 -> 1D transfer.
__device__ __forceinline__ void tdm_load(uint32_t lds_addr, const void* gaddr,
                                         uint32_t tile_d0, uint32_t tile_d1,
                                         uint32_t stride0_units) {
  const uint64_t ga = (uint64_t)(uintptr_t)gaddr;
  // group0: count=1 | lds_addr | global_addr[56:0] | type=2 ("image")
  u32x4 g0 = {1u, lds_addr, (uint32_t)ga,
              (uint32_t)((ga >> 32) & 0x01FFFFFFu) | 0x80000000u};
  // group1: data_size=3(8B); tensor dims == tile dims (no OOB); stride0.
  i32x8 g1 = {(int)(3u << 16),
              (int)((tile_d0 & 0xFFFFu) << 16),                        // tensor_dim0 lo16
              (int)((tile_d0 >> 16) | ((tile_d1 & 0xFFFFu) << 16)),    // td0 hi | td1 lo
              (int)((tile_d1 >> 16) | ((tile_d0 & 0xFFFFu) << 16)),    // td1 hi | tile_dim0
              (int)(tile_d1 & 0xFFFFu),                                // tile_dim1, tile_dim2=0
              (int)stride0_units, 0, 0};
  i32x4 z4 = {0, 0, 0, 0};
#if __clang_major__ >= 23
  i32x8 z8 = {0, 0, 0, 0, 0, 0, 0, 0};
  __builtin_amdgcn_tensor_load_to_lds(g0, g1, z4, z4, z8, 0);
#else
  __builtin_amdgcn_tensor_load_to_lds(g0, g1, z4, z4, 0);
#endif
}

__device__ __forceinline__ void tdm_wait() {
#if __has_builtin(__builtin_amdgcn_s_wait_tensorcnt)
  __builtin_amdgcn_s_wait_tensorcnt((short)0);
#else
  asm volatile("s_wait_tensorcnt 0x0" ::: "memory");
#endif
}
#endif  // HAVE_TDM

#if HAVE_ASYNC
__device__ __forceinline__ void async_b128(const void* g, void* l) {
  // Param type per compiler diagnostic: generic pointer to int __vector(4).
  __builtin_amdgcn_global_load_async_to_lds_b128((v4i_vs*)g, (v4i_vs*)l, 0, 0);
}
__device__ __forceinline__ void async_wait() {
#if __has_builtin(__builtin_amdgcn_s_wait_asynccnt)
  __builtin_amdgcn_s_wait_asynccnt(0);
#else
  asm volatile("s_wait_asynccnt 0x0" ::: "memory");
#endif
}
#endif  // HAVE_ASYNC

// Stage one contiguous 8 KB block (global -> LDS). Caller must __syncthreads().
__device__ __forceinline__ void stage_contig_8kb(void* dst_lds, const void* src_global) {
#if HAVE_TDM
  if ((threadIdx.x >> 5) == 0) {
    tdm_load(lds_off_of(dst_lds), src_global, /*tile_d0=*/1024, /*tile_d1=*/0,
             /*stride0=*/1024);
    tdm_wait();
  }
#elif HAVE_ASYNC
  {
    const char* s = (const char*)src_global + threadIdx.x * 16;
    char* d = (char*)dst_lds + threadIdx.x * 16;
    async_b128(s, d);
    async_b128(s + 4096, d + 4096);
    async_wait();
  }
#else
  {
    const uint32_t* src = (const uint32_t*)src_global;
    uint32_t* dst = (uint32_t*)dst_lds;
    for (int i = threadIdx.x; i < 2048; i += 256) dst[i] = src[i];
  }
#endif
}

// Stage 64 rows x 128 bytes from strided global rows into packed LDS tile.
__device__ __forceinline__ void stage_rows_64x128(void* dst_lds, const void* src_global,
                                                  int row_stride_bytes) {
#if HAVE_TDM
  if ((threadIdx.x >> 5) == 0) {
    tdm_load(lds_off_of(dst_lds), src_global, /*tile_d0=*/16, /*tile_d1=*/64,
             /*stride0=*/(uint32_t)(row_stride_bytes >> 3));
    tdm_wait();
  }
#elif HAVE_ASYNC
  for (int i = threadIdx.x; i < 512; i += 256) {
    const int r = i >> 3, c = i & 7;
    async_b128((const char*)src_global + r * row_stride_bytes + c * 16,
               (char*)dst_lds + r * 128 + c * 16);
  }
  async_wait();
#else
  {
    uint32_t* dst = (uint32_t*)dst_lds;
    const char* src = (const char*)src_global;
    for (int i = threadIdx.x; i < 2048; i += 256) {
      const int r = i >> 5, w = i & 31;
      dst[r * 32 + w] = ((const uint32_t*)(src + r * row_stride_bytes))[w];
    }
  }
#endif
}

// ---------------------------------------------------------------------------
// WMMA helpers (CDNA5 wave32, v_wmma_f32_16x16x32_bf16)
// Fragment layouts per CDNA5 ISA 7.12.2:
//   A (16x32 bf16): lane L -> row M = L&15; half = L>>4;
//       VGPR j<4 : K = half*8 + 2j (+pair)   VGPR j>=4: K = 16 + half*8 + 2(j-4)
//   B (32x16 bf16): lane L -> col N = L&15; half = L>>4; element e<8: K = half*8 + e, e>=8: K = 16 + half*8 + (e-8)
//   C/D (16x16 f32): lane L -> col N = L&15; VGPR i -> row M = i + 8*(L>>4)
// ---------------------------------------------------------------------------
__device__ __forceinline__ v8f wmma_bf16(v16bf a, v16bf b, v8f c) {
  return __builtin_amdgcn_wmma_f32_16x16x32_bf16(
      /*neg_a=*/false, a, /*neg_b=*/false, b,
      /*c_mod=*/(short)0, c, /*reuse_a=*/false, /*reuse_b=*/false);
}

__device__ __forceinline__ v16bf frag_a_bf16(const __bf16* base, int stride,
                                             int row_base, int k_base) {
  const int lane = threadIdx.x & 31;
  const int hl   = lane >> 4;
  const __bf16* p = base + (row_base + (lane & 15)) * stride + k_base;
  v16bf a;
#pragma unroll
  for (int j = 0; j < 8; ++j) {
    const int k = (j < 4) ? (hl * 8 + 2 * j) : (16 + hl * 8 + 2 * (j - 4));
    a[2 * j]     = p[k];
    a[2 * j + 1] = p[k + 1];
  }
  return a;
}

__device__ __forceinline__ v16bf frag_a_f32(const float* base, int stride,
                                            int row_base, int k_base) {
  const int lane = threadIdx.x & 31;
  const int hl   = lane >> 4;
  const float* p = base + (row_base + (lane & 15)) * stride + k_base;
  v16bf a;
#pragma unroll
  for (int j = 0; j < 8; ++j) {
    const int k = (j < 4) ? (hl * 8 + 2 * j) : (16 + hl * 8 + 2 * (j - 4));
    a[2 * j]     = (__bf16)p[k];
    a[2 * j + 1] = (__bf16)p[k + 1];
  }
  return a;
}

__device__ __forceinline__ v16bf frag_b_row(const __bf16* base, int stride,
                                            int k_base, int col_base) {
  const int lane = threadIdx.x & 31;
  const int hl   = lane >> 4;
  const __bf16* p = base + k_base * stride + col_base + (lane & 15);
  v16bf b;
#pragma unroll
  for (int e = 0; e < 16; ++e) {
    const int k = (e < 8) ? (hl * 8 + e) : (16 + hl * 8 + (e - 8));
    b[e] = p[k * stride];
  }
  return b;
}

__device__ __forceinline__ v16bf frag_b_colmajor(const __bf16* base, int stride,
                                                 int k_base, int col_base) {
  const int lane = threadIdx.x & 31;
  const int hl   = lane >> 4;
  const __bf16* p = base + (col_base + (lane & 15)) * stride + k_base;
  v16bf b;
#pragma unroll
  for (int e = 0; e < 16; ++e) {
    const int k = (e < 8) ? (hl * 8 + e) : (16 + hl * 8 + (e - 8));
    b[e] = p[k];
  }
  return b;
}

// ---------------------------------------------------------------------------
// Kernel 1: qkv = x @ w_qkv, emitted as bf16 in head-major layout
// ---------------------------------------------------------------------------
__global__ void qkv_kernel(const float* __restrict__ x,
                           const float* __restrict__ w_qkv,
                           __bf16* __restrict__ qkv_ws) {
  extern __shared__ char smem[];
  __bf16* xs  = (__bf16*)smem;            // 64 x 64   (8 KB)
  __bf16* wsb = xs + 64 * 64;             // 64 x 256  (32 KB)

  const int row0 = blockIdx.x * 64;
  const int col0 = blockIdx.y * 256;
  const int tid  = threadIdx.x;

  for (int i = tid; i < 64 * 64; i += 256) {
    const int r = i >> 6, k = i & 63;
    xs[i] = (__bf16)x[(row0 + r) * 64 + k];
  }
  for (int i = tid; i < 64 * 256; i += 256) {
    const int k = i >> 8, c = i & 255;
    wsb[i] = (__bf16)w_qkv[k * 1536 + col0 + c];
  }
  __syncthreads();

  const int wave = tid >> 5;
  const int lane = tid & 31;
  const int hl   = lane >> 4;
  const int row_sub = wave >> 1;          // 0..3
  const int chalf   = wave & 1;           // 0..1 -> 128-col half

  v8f acc[8];
#pragma unroll
  for (int t = 0; t < 8; ++t)
#pragma unroll
    for (int i = 0; i < 8; ++i) acc[t][i] = 0.0f;

#pragma unroll
  for (int ks = 0; ks < 2; ++ks) {
    const v16bf a = frag_a_bf16(xs, 64, row_sub * 16, ks * 32);
#pragma unroll
    for (int cs = 0; cs < 8; ++cs) {
      const v16bf b = frag_b_row(wsb, 256, ks * 32, chalf * 128 + cs * 16);
      acc[cs] = wmma_bf16(a, b, acc[cs]);
    }
  }

#pragma unroll
  for (int cs = 0; cs < 8; ++cs) {
    const int col   = col0 + chalf * 128 + cs * 16 + (lane & 15);
    const int which = col >> 9;
    const int h     = (col >> 6) & 7;
    const int d     = col & 63;
#pragma unroll
    for (int i = 0; i < 8; ++i) {
      const int R = row0 + row_sub * 16 + i + 8 * hl;
      const int b = R >> 10, n = R & 1023;
      qkv_ws[which * (B_ * H_ * N_ * 64) + (((b * H_ + h) * N_ + n) << 6) + d] =
          (__bf16)acc[cs][i];
    }
  }
}

// ---------------------------------------------------------------------------
// Kernel 2: gamma = x @ w_gamma + b_gamma  (8192 x 2, fp32)
// ---------------------------------------------------------------------------
__global__ void gamma_kernel(const float* __restrict__ x,
                             const float* __restrict__ w_gamma,
                             const float* __restrict__ b_gamma,
                             float* __restrict__ gamma_ws) {
  const int R = blockIdx.x * blockDim.x + threadIdx.x;
  if (R >= ROWS_) return;
  float s0 = b_gamma[0], s1 = b_gamma[1];
#pragma unroll 8
  for (int k = 0; k < 64; ++k) {
    const float xv = x[R * 64 + k];
    s0 += xv * w_gamma[k * 2 + 0];
    s1 += xv * w_gamma[k * 2 + 1];
  }
  gamma_ws[R * 2 + 0] = s0;
  gamma_ws[R * 2 + 1] = s1;
}

// ---------------------------------------------------------------------------
// Kernel 3: fused attention. One block per (b, h, 64-row tile).
// Full 64x1024 score block lives in LDS (256 KB of the 320 KB WGP LDS).
// ---------------------------------------------------------------------------
__global__ void attn_kernel(const __bf16* __restrict__ q_ws,
                            const __bf16* __restrict__ k_ws,
                            const __bf16* __restrict__ v_ws,
                            const float*  __restrict__ adj,
                            const float*  __restrict__ gamma_ws,
                            __bf16* __restrict__ o_ws) {
  extern __shared__ char smem[];
  float*  S   = (float*)smem;                       // 64 x 1024 fp32 (256 KB)
  __bf16* qt  = (__bf16*)(smem + 64 * 1024 * 4);    // 64 x 64  (8 KB)
  __bf16* kv  = qt + 64 * 64;                       // 64 x 64  (8 KB)
  float*  g0s = (float*)(kv + 64 * 64);             // 64
  float*  g1s = g0s + 64;                           // 64

  const int bh = blockIdx.x >> 4;                   // b*H + h
  const int rt = blockIdx.x & 15;
  const int b  = bh >> 3;
  const int h  = bh & 7;
  const int row0 = rt * 64;

  const __bf16* qh = q_ws + bh * (N_ * 64);
  const __bf16* kh = k_ws + bh * (N_ * 64);
  const __bf16* vh = v_ws + bh * (N_ * 64);

  const int tid  = threadIdx.x;
  const int wave = tid >> 5;
  const int lane = tid & 31;
  const int hl   = lane >> 4;
  const int row_sub = wave & 3;                     // 0..3
  const int cs0     = (wave >> 2) * 2;              // {0,2}

  stage_contig_8kb(qt, qh + row0 * 64);
  if (tid < 64) {
    const int n = row0 + tid;
    g0s[tid] = gamma_ws[(b * N_ + n) * 2 + 0];
    g1s[tid] = gamma_ws[(b * N_ + n) * 2 + 1];
  }
  __syncthreads();

  // Q-tile A fragments are invariant over all 16 K/V chunks: build once.
  v16bf a_q[2];
#pragma unroll
  for (int ds = 0; ds < 2; ++ds)
    a_q[ds] = frag_a_bf16(qt, 64, row_sub * 16, ds * 32);

  // ---- Phase 1: S = g0 * scale * (Q K^T) + g1 * bias ----------------------
  for (int mc = 0; mc < 16; ++mc) {
    const int mb = mc * 64;
    stage_contig_8kb(kv, kh + mb * 64);
    if (mc + 1 < 16) __builtin_prefetch(kh + (mb + 64) * 64, 0, 0);
    __syncthreads();

#pragma unroll
    for (int t = 0; t < 2; ++t) {
      const int cs = cs0 + t;
      v8f acc;
#pragma unroll
      for (int i = 0; i < 8; ++i) acc[i] = 0.0f;
#pragma unroll
      for (int ds = 0; ds < 2; ++ds) {
        const v16bf bb = frag_b_colmajor(kv, 64, ds * 32, cs * 16);
        acc = wmma_bf16(a_q[ds], bb, acc);
      }
      const int m = mb + cs * 16 + (lane & 15);
#pragma unroll
      for (int i = 0; i < 8; ++i) {
        const int r = row_sub * 16 + i + 8 * hl;
        const int n = row0 + r;
        float bias;
        if (n == 0)      bias = (m == 0) ? 1.0f : 0.0f;
        else if (m == 0) bias = 0.0f;
        else             bias = adj[(b * 1023 + (n - 1)) * 1023 + (m - 1)];
        S[r * 1024 + m] = g0s[r] * acc[i] * SCALE_ + g1s[r] * bias;
      }
    }
    __syncthreads();
  }

  // ---- Phase 2: exact row softmax in LDS (wave32 shuffles) ----------------
#pragma unroll
  for (int qr = 0; qr < 8; ++qr) {
    float* row = S + (wave * 8 + qr) * 1024;
    float mx = -3.4e38f;
    for (int j = lane; j < 1024; j += 32) mx = fmaxf(mx, row[j]);
#pragma unroll
    for (int off = 16; off > 0; off >>= 1)
      mx = fmaxf(mx, __shfl_xor(mx, off, 32));
    float sum = 0.0f;
    for (int j = lane; j < 1024; j += 32) {
      const float e = __expf(row[j] - mx);
      row[j] = e;
      sum += e;
    }
#pragma unroll
    for (int off = 16; off > 0; off >>= 1)
      sum += __shfl_xor(sum, off, 32);
    const float inv = 1.0f / sum;
    for (int j = lane; j < 1024; j += 32) row[j] *= inv;
  }
  __syncthreads();

  // ---- Phase 3: O = P @ V -------------------------------------------------
  v8f oacc[2];
#pragma unroll
  for (int t = 0; t < 2; ++t)
#pragma unroll
    for (int i = 0; i < 8; ++i) oacc[t][i] = 0.0f;

  for (int mc = 0; mc < 16; ++mc) {
    const int mb = mc * 64;
    stage_contig_8kb(kv, vh + mb * 64);
    if (mc + 1 < 16) __builtin_prefetch(vh + (mb + 64) * 64, 0, 0);
    __syncthreads();

#pragma unroll
    for (int ms = 0; ms < 2; ++ms) {
      const v16bf a = frag_a_f32(S, 1024, row_sub * 16, mb + ms * 32);
#pragma unroll
      for (int t = 0; t < 2; ++t) {
        const v16bf bb = frag_b_row(kv, 64, ms * 32, (cs0 + t) * 16);
        oacc[t] = wmma_bf16(a, bb, oacc[t]);
      }
    }
    __syncthreads();
  }

#pragma unroll
  for (int t = 0; t < 2; ++t) {
    const int d = (cs0 + t) * 16 + (lane & 15);
#pragma unroll
    for (int i = 0; i < 8; ++i) {
      const int r = row_sub * 16 + i + 8 * hl;
      const int n = row0 + r;
      o_ws[((b * N_ + n) * H_ + h) * 64 + d] = (__bf16)oacc[t][i];
    }
  }
}

// ---------------------------------------------------------------------------
// Kernel 4: out = O(8192x512) @ w_proj(512x64) + b_proj  (fp32 result)
// ---------------------------------------------------------------------------
__global__ void proj_kernel(const __bf16* __restrict__ o_ws,
                            const float* __restrict__ w_proj,
                            const float* __restrict__ b_proj,
                            float* __restrict__ out) {
  extern __shared__ char smem[];
  __bf16* wp = (__bf16*)smem;              // 512 x 64 (64 KB)
  __bf16* ot = wp + 512 * 64;              // 64 x 64  (8 KB)

  const int row0 = blockIdx.x * 64;
  const int tid  = threadIdx.x;
  const int wave = tid >> 5;
  const int lane = tid & 31;
  const int hl   = lane >> 4;
  const int row_sub = wave & 3;
  const int cs0     = (wave >> 2) * 2;

  for (int i = tid; i < 512 * 64; i += 256) wp[i] = (__bf16)w_proj[i];
  __syncthreads();

  v8f acc[2];
#pragma unroll
  for (int t = 0; t < 2; ++t)
#pragma unroll
    for (int i = 0; i < 8; ++i) acc[t][i] = 0.0f;

  for (int kc = 0; kc < 8; ++kc) {
    const int kb = kc * 64;
    stage_rows_64x128(ot, (const char*)o_ws + (size_t)row0 * 1024 + kb * 2,
                      /*row_stride_bytes=*/1024);
    __syncthreads();
#pragma unroll
    for (int ks = 0; ks < 2; ++ks) {
      const v16bf a = frag_a_bf16(ot, 64, row_sub * 16, ks * 32);
#pragma unroll
      for (int t = 0; t < 2; ++t) {
        const v16bf bb = frag_b_row(wp, 64, kb + ks * 32, (cs0 + t) * 16);
        acc[t] = wmma_bf16(a, bb, acc[t]);
      }
    }
    __syncthreads();
  }

#pragma unroll
  for (int t = 0; t < 2; ++t) {
    const int col = (cs0 + t) * 16 + (lane & 15);
    const float bp = b_proj[col];
#pragma unroll
    for (int i = 0; i < 8; ++i) {
      const int R = row0 + row_sub * 16 + i + 8 * hl;
      out[R * 64 + col] = acc[t][i] + bp;
    }
  }
}

// ---------------------------------------------------------------------------
// Host launcher
// ---------------------------------------------------------------------------
extern "C" void kernel_launch(void* const* d_in, const int* in_sizes, int n_in,
                              void* d_out, int out_size, void* d_ws, size_t ws_size,
                              hipStream_t stream) {
  const float* x       = (const float*)d_in[0];
  const float* adj     = (const float*)d_in[1];
  const float* w_qkv   = (const float*)d_in[2];
  const float* w_proj  = (const float*)d_in[3];
  const float* b_proj  = (const float*)d_in[4];
  const float* w_gamma = (const float*)d_in[5];
  const float* b_gamma = (const float*)d_in[6];
  float* out = (float*)d_out;

  // Workspace layout (bytes):
  //   [0, 25165824)        q/k/v bf16   (3 * 8*8*1024*64 * 2)
  //   [25165824, 33554432) O bf16       (8*1024*512 * 2)
  //   [33554432, 33619968) gamma fp32   (8*1024*2 * 4)
  char* ws = (char*)d_ws;
  __bf16* qkv_ws = (__bf16*)ws;
  __bf16* q_ws   = qkv_ws;
  __bf16* k_ws   = qkv_ws + (size_t)B_ * H_ * N_ * 64;
  __bf16* v_ws   = qkv_ws + (size_t)2 * B_ * H_ * N_ * 64;
  __bf16* o_ws   = (__bf16*)(ws + 25165824u);
  float*  g_ws   = (float*)(ws + 33554432u);

  const size_t smem_qkv  = (64 * 64 + 64 * 256) * sizeof(__bf16);           // 40 KB
  const size_t smem_attn = 64 * 1024 * 4 + 2 * 64 * 64 * 2 + 128 * 4;       // ~272.5 KB
  const size_t smem_proj = (512 * 64 + 64 * 64) * sizeof(__bf16);           // 72 KB

  (void)hipFuncSetAttribute((const void*)attn_kernel,
                            hipFuncAttributeMaxDynamicSharedMemorySize, (int)smem_attn);
  (void)hipFuncSetAttribute((const void*)proj_kernel,
                            hipFuncAttributeMaxDynamicSharedMemorySize, (int)smem_proj);

  qkv_kernel<<<dim3(ROWS_ / 64, 1536 / 256), 256, smem_qkv, stream>>>(x, w_qkv, qkv_ws);
  gamma_kernel<<<ROWS_ / 256, 256, 0, stream>>>(x, w_gamma, b_gamma, g_ws);
  attn_kernel<<<B_ * H_ * (N_ / 64), 256, smem_attn, stream>>>(
      q_ws, k_ws, v_ws, adj, g_ws, o_ws);
  proj_kernel<<<ROWS_ / 64, 256, smem_proj, stream>>>(o_ws, w_proj, b_proj, out);
}
